// OmegaExpert_65773129171180
// MI455X (gfx1250) — 8-bit matrix operand order repaired and checked by reading, compiled, not run on hardware
//
#include <hip/hip_runtime.h>
#include <hip/hip_bf16.h>

typedef __attribute__((ext_vector_type(8))) int v8i;

#define EPSF 1e-5f

__device__ __forceinline__ void async_b128(const char* g, char* l) {
  unsigned lv = (unsigned)(unsigned long long)l;
  asm volatile("global_load_async_to_lds_b128 %0, %1, off"
               :
               : "v"(lv), "v"((unsigned long long)g)
               : "memory");
}
__device__ __forceinline__ void wait_async_le(int) = delete;
__device__ __forceinline__ void wait_async_le4() {
  asm volatile("s_wait_asynccnt 0x4" ::: "memory");
}
__device__ __forceinline__ void wait_async_0() {
  asm volatile("s_wait_asynccnt 0x0" ::: "memory");
}

__global__ void k_zero_u32(unsigned int* p, int n) {
  int i = blockIdx.x * blockDim.x + threadIdx.x;
  if (i < n) p[i] = 0u;
}

__global__ void k_abs_partial(const float* __restrict__ w, size_t n,
                              float* __restrict__ part) {
  __shared__ float sm[256];
  float s = 0.f;
  size_t stride = (size_t)gridDim.x * blockDim.x;
  for (size_t i = (size_t)blockIdx.x * blockDim.x + threadIdx.x; i < n; i += stride)
    s += fabsf(w[i]);
  sm[threadIdx.x] = s;
  __syncthreads();
  for (int o = 128; o > 0; o >>= 1) {
    if ((int)threadIdx.x < o) sm[threadIdx.x] += sm[threadIdx.x + o];
    __syncthreads();
  }
  if (threadIdx.x == 0) part[blockIdx.x] = sm[0];
}

__global__ void k_abs_final(const float* __restrict__ part, int nparts,
                            float inv_n, float* __restrict__ wscale) {
  __shared__ float sm[256];
  float s = 0.f;
  for (int i = threadIdx.x; i < nparts; i += 256) s += part[i];
  sm[threadIdx.x] = s;
  __syncthreads();
  for (int o = 128; o > 0; o >>= 1) {
    if ((int)threadIdx.x < o) sm[threadIdx.x] += sm[threadIdx.x + o];
    __syncthreads();
  }
  if (threadIdx.x == 0) wscale[0] = fmaxf(sm[0] * inv_n, EPSF);
}

__global__ void k_quant_w(const float4* __restrict__ w4, size_t n4,
                          const float* __restrict__ wscale,
                          char4* __restrict__ wq4) {
  size_t i = (size_t)blockIdx.x * blockDim.x + threadIdx.x;
  if (i >= n4) return;
  float inv = 1.0f / wscale[0];
  float4 v = w4[i];
  int a = (int)rintf(v.x * inv);
  int b = (int)rintf(v.y * inv);
  int c = (int)rintf(v.z * inv);
  int d = (int)rintf(v.w * inv);
  char4 o;
  o.x = (char)max(-1, min(1, a));
  o.y = (char)max(-1, min(1, b));
  o.z = (char)max(-1, min(1, c));
  o.w = (char)max(-1, min(1, d));
  wq4[i] = o;
}

__global__ void k_quant_x(const float* __restrict__ x, int K,
                          char* __restrict__ xq, float* __restrict__ inv_sx) {
  __shared__ float sm[256];
  int row = blockIdx.x;
  const float* xr = x + (size_t)row * K;
  float m = 0.f;
  for (int i = threadIdx.x; i < K; i += 256) m = fmaxf(m, fabsf(xr[i]));
  sm[threadIdx.x] = m;
  __syncthreads();
  for (int o = 128; o > 0; o >>= 1) {
    if ((int)threadIdx.x < o)
      sm[threadIdx.x] = fmaxf(sm[threadIdx.x], sm[threadIdx.x + o]);
    __syncthreads();
  }
  float amax = fmaxf(sm[0], EPSF);
  float scale = 127.0f / amax;
  if (threadIdx.x == 0) inv_sx[row] = amax / 127.0f;
  char* out = xq + (size_t)row * K;
  for (int i = threadIdx.x; i < K; i += 256) {
    int q = (int)rintf(xr[i] * scale);
    out[i] = (char)max(-128, min(127, q));
  }
}

__global__ void k_quant_h(const float* __restrict__ H, int N,
                          const unsigned int* __restrict__ rowmax,
                          char* __restrict__ Hq, float* __restrict__ inv_sh) {
  int row = blockIdx.x;
  float amax = fmaxf(__uint_as_float(rowmax[row]), EPSF);
  float scale = 127.0f / amax;
  if (threadIdx.x == 0) inv_sh[row] = amax / 127.0f;
  const float* hr = H + (size_t)row * N;
  char* oq = Hq + (size_t)row * N;
  for (int i = threadIdx.x; i < N; i += blockDim.x) {
    int q = (int)rintf(hr[i] * scale);
    oq[i] = (char)max(-128, min(127, q));
  }
}

__global__ __launch_bounds__(256) void k_gemm_gateup(
    const char* __restrict__ Aq,
    const char* __restrict__ Bgq,
    const char* __restrict__ Buq,
    const float* __restrict__ inv_sx,
    const float* __restrict__ wgs,
    const float* __restrict__ wus,
    float* __restrict__ H,
    unsigned int* __restrict__ rowmax,
    int M, int N, int K) {
  __shared__ char lA[3][128 * 64];
  __shared__ char lBg[3][64 * 64];
  __shared__ char lBu[3][64 * 64];
  __shared__ unsigned int rmax[128];

  const int t = threadIdx.x;
  const int m0 = blockIdx.y * 128;
  const int n0 = blockIdx.x * 64;

  if (t < 128) rmax[t] = 0u;

  const int wave = t >> 5, lane = t & 31;
  const int waveM = wave >> 1;
  const int waveN = wave & 1;
  const int lr = lane & 15, half = lane >> 4;

  v8i accg[2][2], accu[2][2];
  for (int i = 0; i < 2; i++)
    for (int j = 0; j < 2; j++) {
      accg[i][j] = {};
      accu[i][j] = {};
    }

  const int rowA = t >> 1, offA = (t & 1) * 32;
  const int rowB = t >> 2, offB = (t & 3) * 16;

  auto issue_stage = [&](int kb, int s) {
    const int k0 = kb * 64;
    const char* ga = Aq + (size_t)(m0 + rowA) * K + k0 + offA;
    char* la = &lA[s][rowA * 64 + offA];
    async_b128(ga, la);
    async_b128(ga + 16, la + 16);
    async_b128(Bgq + (size_t)(n0 + rowB) * K + k0 + offB,
               &lBg[s][rowB * 64 + offB]);
    async_b128(Buq + (size_t)(n0 + rowB) * K + k0 + offB,
               &lBu[s][rowB * 64 + offB]);
  };

  const int KB = K >> 6;
  issue_stage(0, 0);
  int s = 0, sn = 1;
  for (int kb = 0; kb < KB; ++kb) {
    if (kb + 1 < KB) {
      issue_stage(kb + 1, sn);
      wait_async_le4();
    } else {
      wait_async_0();
    }
    __syncthreads();

    v8i a[2];
    for (int mt = 0; mt < 2; mt++) {
      const char* rp = &lA[s][(waveM * 32 + mt * 16 + lr) * 64 + half * 16];
      union { v8i v; unsigned long long u[4]; } au;
      au.u[0] = *(const unsigned long long*)(rp + 0);
      au.u[1] = *(const unsigned long long*)(rp + 8);
      au.u[2] = *(const unsigned long long*)(rp + 32);
      au.u[3] = *(const unsigned long long*)(rp + 40);
      a[mt] = au.v;
    }
    v8i bg[2], bu[2];
    for (int nt = 0; nt < 2; nt++) {
      const char* rg = &lBg[s][(waveN * 32 + nt * 16 + lr) * 64 + half * 16];
      union { v8i v; uint4 q[2]; } b;
      b.q[0] = *(const uint4*)(rg + 0);
      b.q[1] = *(const uint4*)(rg + 32);
      bg[nt] = b.v;
      const char* ru = &lBu[s][(waveN * 32 + nt * 16 + lr) * 64 + half * 16];
      b.q[0] = *(const uint4*)(ru + 0);
      b.q[1] = *(const uint4*)(ru + 32);
      bu[nt] = b.v;
    }
    for (int mt = 0; mt < 2; mt++)
      for (int nt = 0; nt < 2; nt++) {
        accg[mt][nt] = __builtin_amdgcn_wmma_i32_16x16x64_iu8(
            true, a[mt], true, bg[nt], accg[mt][nt], false, false);
        accu[mt][nt] = __builtin_amdgcn_wmma_i32_16x16x64_iu8(
            true, a[mt], true, bu[nt], accu[mt][nt], false, false);
      }

    s = sn;
    sn = (sn == 2) ? 0 : sn + 1;
  }

  const float gsc = wgs[0], usc = wus[0];
  for (int mt = 0; mt < 2; mt++) {
    for (int v = 0; v < 8; v++) {
      int mloc = waveM * 32 + mt * 16 + v + half * 8;
      int m = m0 + mloc;
      float dx = inv_sx[m];
      float lm = 0.f;
      for (int nt = 0; nt < 2; nt++) {
        float g = (float)accg[mt][nt][v] * dx * gsc;
        float u = (float)accu[mt][nt][v] * dx * usc;
        float gr = fmaxf(g, 0.f);
        float h = gr * gr * u;
        int n = n0 + waveN * 32 + nt * 16 + lr;
        H[(size_t)m * N + n] = h;
        lm = fmaxf(lm, fabsf(h));
      }
      atomicMax(&rmax[mloc], __float_as_uint(lm));
    }
  }
  __syncthreads();
  if (t < 128) atomicMax(&rowmax[m0 + t], rmax[t]);
}

__global__ __launch_bounds__(256) void k_gemm_down(
    const char* __restrict__ Aq,
    const char* __restrict__ Bq,
    const float* __restrict__ inv_sh,
    const float* __restrict__ wds,
    float* __restrict__ Out,
    int M, int N, int K) {
  __shared__ char lA[3][128 * 64];
  __shared__ char lB[3][128 * 64];

  const int t = threadIdx.x;
  const int m0 = blockIdx.y * 128;
  const int n0 = blockIdx.x * 128;
  const int wave = t >> 5, lane = t & 31;
  const int waveM = wave >> 2;
  const int waveN = wave & 3;
  const int lr = lane & 15, half = lane >> 4;

  v8i acc[4][2];
  for (int i = 0; i < 4; i++)
    for (int j = 0; j < 2; j++) acc[i][j] = {};

  const int rowT = t >> 1, offT = (t & 1) * 32;

  auto issue_stage = [&](int kb, int s) {
    const int k0 = kb * 64;
    const char* ga = Aq + (size_t)(m0 + rowT) * K + k0 + offT;
    char* la = &lA[s][rowT * 64 + offT];
    async_b128(ga, la);
    async_b128(ga + 16, la + 16);
    const char* gb = Bq + (size_t)(n0 + rowT) * K + k0 + offT;
    char* lb = &lB[s][rowT * 64 + offT];
    async_b128(gb, lb);
    async_b128(gb + 16, lb + 16);
  };

  const int KB = K >> 6;
  issue_stage(0, 0);
  int s = 0, sn = 1;
  for (int kb = 0; kb < KB; ++kb) {
    if (kb + 1 < KB) {
      issue_stage(kb + 1, sn);
      wait_async_le4();
    } else {
      wait_async_0();
    }
    __syncthreads();

    v8i b[2];
    for (int nt = 0; nt < 2; nt++) {
      const char* rb = &lB[s][(waveN * 32 + nt * 16 + lr) * 64 + half * 16];
      union { v8i v; uint4 q[2]; } bu;
      bu.q[0] = *(const uint4*)(rb + 0);
      bu.q[1] = *(const uint4*)(rb + 32);
      b[nt] = bu.v;
    }
    for (int mt = 0; mt < 4; mt++) {
      const char* rp = &lA[s][(waveM * 64 + mt * 16 + lr) * 64 + half * 16];
      union { v8i v; unsigned long long u[4]; } au;
      au.u[0] = *(const unsigned long long*)(rp + 0);
      au.u[1] = *(const unsigned long long*)(rp + 8);
      au.u[2] = *(const unsigned long long*)(rp + 32);
      au.u[3] = *(const unsigned long long*)(rp + 40);
      for (int nt = 0; nt < 2; nt++)
        acc[mt][nt] = __builtin_amdgcn_wmma_i32_16x16x64_iu8(
            true, au.v, true, b[nt], acc[mt][nt], false, false);
    }

    s = sn;
    sn = (sn == 2) ? 0 : sn + 1;
  }

  const float wsc = wds[0];
  for (int mt = 0; mt < 4; mt++) {
    for (int v = 0; v < 8; v++) {
      int m = m0 + waveM * 64 + mt * 16 + v + half * 8;
      float d = inv_sh[m] * wsc;
      for (int nt = 0; nt < 2; nt++) {
        int n = n0 + waveN * 32 + nt * 16 + lr;
        Out[(size_t)m * N + n] = (float)acc[mt][nt][v] * d;
      }
    }
  }
}

extern "C" void kernel_launch(void* const* d_in, const int* in_sizes, int n_in,
                              void* d_out, int out_size, void* d_ws,
                              size_t ws_size, hipStream_t stream) {
  const float* x      = (const float*)d_in[0];
  const float* w_gate = (const float*)d_in[1];
  const float* w_up   = (const float*)d_in[2];
  const float* w_down = (const float*)d_in[3];

  const int Mtok = 4096;
  const int Hh   = 4096;
  const int Ii   = 11008;

  char* ws = (char*)d_ws;
  size_t off = 0;
  auto alloc = [&](size_t bytes) -> char* {
    char* p = ws + off;
    off = (off + bytes + 255) & ~(size_t)255;
    return p;
  };

  char* xq   = alloc((size_t)Mtok * Hh);
  char* wqg  = alloc((size_t)Ii * Hh);
  char* wqu  = alloc((size_t)Ii * Hh);
  char* wqd  = alloc((size_t)Hh * Ii);
  char* hq   = alloc((size_t)Mtok * Ii);
  float* hbuf = (float*)alloc((size_t)Mtok * Ii * sizeof(float));
  float* inv_sx = (float*)alloc((size_t)Mtok * sizeof(float));
  float* inv_sh = (float*)alloc((size_t)Mtok * sizeof(float));
  unsigned int* rowmax = (unsigned int*)alloc((size_t)Mtok * sizeof(unsigned int));
  float* wsc  = (float*)alloc(4 * sizeof(float));
  float* part = (float*)alloc(3 * 1024 * sizeof(float));

  k_zero_u32<<<(Mtok + 255) / 256, 256, 0, stream>>>(rowmax, Mtok);

  const size_t nw = (size_t)Ii * Hh;
  k_abs_partial<<<1024, 256, 0, stream>>>(w_gate, nw, part);
  k_abs_final<<<1, 256, 0, stream>>>(part, 1024, 1.0f / (float)nw, wsc + 0);
  k_abs_partial<<<1024, 256, 0, stream>>>(w_up, nw, part + 1024);
  k_abs_final<<<1, 256, 0, stream>>>(part + 1024, 1024, 1.0f / (float)nw, wsc + 1);
  k_abs_partial<<<1024, 256, 0, stream>>>(w_down, nw, part + 2048);
  k_abs_final<<<1, 256, 0, stream>>>(part + 2048, 1024, 1.0f / (float)nw, wsc + 2);

  const size_t n4 = nw / 4;
  const int qb = (int)((n4 + 255) / 256);
  k_quant_w<<<qb, 256, 0, stream>>>((const float4*)w_gate, n4, wsc + 0, (char4*)wqg);
  k_quant_w<<<qb, 256, 0, stream>>>((const float4*)w_up,   n4, wsc + 1, (char4*)wqu);
  k_quant_w<<<qb, 256, 0, stream>>>((const float4*)w_down, n4, wsc + 2, (char4*)wqd);

  k_quant_x<<<Mtok, 256, 0, stream>>>(x, Hh, xq, inv_sx);

  dim3 g1(Ii / 64, Mtok / 128);
  k_gemm_gateup<<<g1, 256, 0, stream>>>(xq, wqg, wqu, inv_sx, wsc + 0, wsc + 1,
                                        hbuf, rowmax, Mtok, Ii, Hh);

  k_quant_h<<<Mtok, 256, 0, stream>>>(hbuf, Ii, rowmax, hq, inv_sh);

  dim3 g2(Hh / 128, Mtok / 128);
  k_gemm_down<<<g2, 256, 0, stream>>>(hq, wqd, inv_sh, wsc + 2, (float*)d_out,
                                      Mtok, Hh, Ii);
}
